// BaseMoEModel_69226282877354
// MI455X (gfx1250) — compile-verified, matrix operand order checked by reading, not run on hardware
//
#include <hip/hip_runtime.h>
#include <math.h>

// ---------------- types ----------------
typedef __bf16 bf16;
typedef __attribute__((ext_vector_type(16))) __bf16 v16bf;
typedef __attribute__((ext_vector_type(8)))  __bf16 bf16x8;
typedef __attribute__((ext_vector_type(8)))  float  v8f;

union FragU { v16bf v; bf16x8 h[2]; };

// Low 32 bits of a generic pointer into LDS are the LDS byte offset (ISA 10.2).
__device__ __forceinline__ unsigned lds_off(const void* p) {
  return (unsigned)(unsigned long long)p;
}

// ---------------- WMMA GEMM: C = A(bf16, MxK) @ Bt(bf16, NxK)^T ----------------
// Block: 128 threads = 4 waves (2x2). Block tile 128x128; wave tile 64x64 = 4x4 wmma tiles.
// Triple-buffered LDS fed by gfx1250 async global->LDS loads; batch of 8 async
// instructions per tile lets us wait "asynccnt <= 8" so the newest prefetch
// stays in flight while the previous one is consumed. Last two K-steps are
// peeled so the steady-state loop is branch-free.
template<bool HAS_BIAS, bool GELU_ACT, bool OUT_BF16, bool ACCUM, bool ROWSCALE>
__global__ __launch_bounds__(128) void gemm_bf16(
    const bf16* __restrict__ A, const bf16* __restrict__ Bt,
    const float* __restrict__ bias, const float* __restrict__ rowscale, int rs_stride,
    float* __restrict__ Cf, bf16* __restrict__ Cb,
    int M, int N, int K)
{
  constexpr int LDSS = 40;                       // 32 + 8 pad (bf16 elems)
  __shared__ bf16 As[3][128 * LDSS];
  __shared__ bf16 Bs[3][128 * LDSS];

  const int tid  = threadIdx.x;
  const int wave = tid >> 5;
  const int lane = tid & 31;
  const int half = lane >> 4;                    // WMMA K-half select
  const int l15  = lane & 15;
  const int bm = blockIdx.y * 128, bn = blockIdx.x * 128;
  const int wm = (wave >> 1) * 64;               // 0 or 64
  const int wn = (wave & 1) * 64;                // 0 or 64

  v8f acc[4][4];
  const v8f zacc = {0.f,0.f,0.f,0.f,0.f,0.f,0.f,0.f};
  #pragma unroll
  for (int i = 0; i < 4; ++i)
    #pragma unroll
    for (int j = 0; j < 4; ++j) acc[i][j] = zacc;

  // loader: thread t owns tile row t (64 bytes of K per step) for both A and Bt
  const bf16* aptr = A  + (size_t)(bm + tid) * K;
  const bf16* bptr = Bt + (size_t)(bn + tid) * K;
  unsigned a_lds[3], b_lds[3];
  #pragma unroll
  for (int i = 0; i < 3; ++i) {
    a_lds[i] = lds_off(&As[i][tid * LDSS]);
    b_lds[i] = lds_off(&Bs[i][tid * LDSS]);
  }

  const int k0ofs = half * 8;                    // bf16 elems
  const int k1ofs = 16 + half * 8;

  auto prefetch = [&](int buf, int k0) {
    unsigned long long ga = (unsigned long long)(aptr + k0);
    unsigned long long gb = (unsigned long long)(bptr + k0);
    asm volatile(
      "global_load_async_to_lds_b128 %0, %2, off\n\t"
      "global_load_async_to_lds_b128 %0, %2, off offset:16\n\t"
      "global_load_async_to_lds_b128 %0, %2, off offset:32\n\t"
      "global_load_async_to_lds_b128 %0, %2, off offset:48\n\t"
      "global_load_async_to_lds_b128 %1, %3, off\n\t"
      "global_load_async_to_lds_b128 %1, %3, off offset:16\n\t"
      "global_load_async_to_lds_b128 %1, %3, off offset:32\n\t"
      "global_load_async_to_lds_b128 %1, %3, off offset:48"
      :: "v"(a_lds[buf]), "v"(b_lds[buf]), "v"(ga), "v"(gb)
      : "memory");
  };

  auto compute = [&](int buf) {
    FragU bfr[4];
    #pragma unroll
    for (int ni = 0; ni < 4; ++ni) {
      const bf16* p = &Bs[buf][(wn + ni * 16 + l15) * LDSS];
      bfr[ni].h[0] = *(const bf16x8*)(p + k0ofs);
      bfr[ni].h[1] = *(const bf16x8*)(p + k1ofs);
    }
    #pragma unroll
    for (int mi = 0; mi < 4; ++mi) {
      FragU af;
      const bf16* p = &As[buf][(wm + mi * 16 + l15) * LDSS];
      af.h[0] = *(const bf16x8*)(p + k0ofs);
      af.h[1] = *(const bf16x8*)(p + k1ofs);
      #pragma unroll
      for (int ni = 0; ni < 4; ++ni)
        acc[mi][ni] = __builtin_amdgcn_wmma_f32_16x16x32_bf16(
            false, af.v, false, bfr[ni].v, (short)0, acc[mi][ni], false, false);
    }
  };

  const int nk = K >> 5;
  prefetch(0, 0);
  if (nk > 1) {
    prefetch(1, 32);
    asm volatile("s_wait_asynccnt 0x8" ::: "memory");  // tile0 landed, tile1 in flight
  } else {
    asm volatile("s_wait_asynccnt 0x0" ::: "memory");
  }
  __syncthreads();

  // steady state: branch-free (always prefetch kk+2, wait for kk+1)
  int cur = 0;
  for (int kk = 0; kk < nk - 2; ++kk) {
    int nb = cur + 2; if (nb >= 3) nb -= 3;      // buffer last read in the previous iter
    prefetch(nb, (kk + 2) << 5);
    compute(cur);
    asm volatile("s_wait_asynccnt 0x8" ::: "memory");  // tile kk+1 landed
    __syncthreads();
    cur = (cur + 1 == 3) ? 0 : cur + 1;
  }
  // kk = nk-2: nothing left to prefetch; drain the last batch
  if (nk >= 2) {
    compute(cur);
    asm volatile("s_wait_asynccnt 0x0" ::: "memory");
    __syncthreads();
    cur = (cur + 1 == 3) ? 0 : cur + 1;
  }
  // kk = nk-1: final tile; no wait/barrier needed after
  compute(cur);

  // epilogue: C 16x16 layout = lane(0-15)=N, vgpr r -> M=r (+8 for lanes 16-31)
  #pragma unroll
  for (int mi = 0; mi < 4; ++mi) {
    const int rbase = bm + wm + mi * 16 + half * 8;
    #pragma unroll
    for (int ni = 0; ni < 4; ++ni) {
      const int c = bn + wn + ni * 16 + l15;
      float bv = 0.f;
      if (HAS_BIAS) bv = bias[c];
      #pragma unroll
      for (int r = 0; r < 8; ++r) {
        const int row = rbase + r;
        float val = acc[mi][ni][r] + bv;
        if (GELU_ACT) val = 0.5f * val * (1.0f + erff(val * 0.70710678118654752f));
        if (ROWSCALE) val *= rowscale[(size_t)row * rs_stride];
        const size_t idx = (size_t)row * N + c;
        if (ACCUM)           Cf[idx] += val;
        else if (OUT_BF16)   Cb[idx]  = (bf16)val;
        else                 Cf[idx]  = val;
      }
    }
  }
}

// ---------------- embedding: x = tok_emb[ids] + pos_emb ----------------
__global__ __launch_bounds__(256) void embed_kernel(
    const int* __restrict__ ids, const float* __restrict__ tok,
    const float* __restrict__ pos, float* __restrict__ x, int S, int D)
{
  const int t = blockIdx.x;
  const int s = t % S;
  const int id = ids[t];
  for (int d = threadIdx.x; d < D; d += 256)
    x[(size_t)t * D + d] = tok[(size_t)id * D + d] + pos[(size_t)s * D + d];
}

// ---------------- layernorm (fp32 in, bf16 out) ----------------
__global__ __launch_bounds__(256) void layernorm_kernel(
    const float* __restrict__ x, const float* __restrict__ w,
    const float* __restrict__ b, bf16* __restrict__ out, int D)
{
  const int t = blockIdx.x, tid = threadIdx.x;
  const int lane = tid & 31, wave = tid >> 5;
  const float* xr = x + (size_t)t * D;
  float v[4];
  #pragma unroll
  for (int i = 0; i < 4; ++i) v[i] = xr[tid + i * 256];
  float s = 0.f, ss = 0.f;
  #pragma unroll
  for (int i = 0; i < 4; ++i) { s += v[i]; ss += v[i] * v[i]; }
  for (int o = 16; o > 0; o >>= 1) { s += __shfl_xor(s, o, 32); ss += __shfl_xor(ss, o, 32); }
  __shared__ float rs[8], rss[8];
  if (lane == 0) { rs[wave] = s; rss[wave] = ss; }
  __syncthreads();
  if (tid == 0) {
    float a = 0.f, c = 0.f;
    #pragma unroll
    for (int i = 0; i < 8; ++i) { a += rs[i]; c += rss[i]; }
    rs[0] = a; rss[0] = c;
  }
  __syncthreads();
  const float mean = rs[0] / D;
  const float var  = rss[0] / D - mean * mean;
  const float rstd = rsqrtf(var + 1e-5f);
  #pragma unroll
  for (int i = 0; i < 4; ++i) {
    const int d = tid + i * 256;
    out[(size_t)t * D + d] = (bf16)(((v[i] - mean) * rstd) * w[d] + b[d]);
  }
}

// ---------------- streaming causal attention (one wave per query row) ----------------
__global__ __launch_bounds__(256) void attn_kernel(
    const bf16* __restrict__ qkv, bf16* __restrict__ o, int S, int Dm, int H)
{
  const int lane = threadIdx.x & 31, wave = threadIdx.x >> 5;
  const int bh = blockIdx.x;
  const int b = bh / H, h = bh % H;
  const int sq = blockIdx.y * 8 + wave;
  const size_t tq = (size_t)b * S + sq;
  const int stride = 3 * Dm;

  __shared__ float qs[8][64];
  const bf16* qrow = qkv + tq * stride + h * 64;
  qs[wave][lane]      = (float)qrow[lane];
  qs[wave][lane + 32] = (float)qrow[lane + 32];
  __syncthreads();

  float m_run = -__builtin_inff(), l_run = 0.f, acc0 = 0.f, acc1 = 0.f;
  const float scale = 0.125f;                    // 1/sqrt(64)

  for (int kb = 0; kb <= sq; kb += 32) {
    const int j = kb + lane;
    float sc = -__builtin_inff();
    if (j <= sq) {
      const bf16* krow = qkv + ((size_t)b * S + j) * stride + Dm + h * 64;
      float d = 0.f;
      #pragma unroll
      for (int c = 0; c < 8; ++c) {
        bf16x8 kv = *(const bf16x8*)(krow + c * 8);
        #pragma unroll
        for (int i = 0; i < 8; ++i) d += qs[wave][c * 8 + i] * (float)kv[i];
      }
      sc = d * scale;
    }
    float mb = sc;
    for (int off = 16; off > 0; off >>= 1) mb = fmaxf(mb, __shfl_xor(mb, off, 32));
    const float m_new = fmaxf(m_run, mb);
    const float p = __expf(sc - m_new);          // 0 for masked lanes
    float lsum = p;
    for (int off = 16; off > 0; off >>= 1) lsum += __shfl_xor(lsum, off, 32);
    const float corr = __expf(m_run - m_new);
    acc0 *= corr; acc1 *= corr;
    l_run = l_run * corr + lsum;
    m_run = m_new;

    const int jmax = (sq - kb < 31) ? (sq - kb) : 31;
    const bf16* vbase = qkv + ((size_t)b * S + kb) * stride + 2 * Dm + h * 64;
    for (int jj = 0; jj <= jmax; ++jj) {
      const float pj = __shfl(p, jj, 32);
      const bf16* vrow = vbase + (size_t)jj * stride;
      acc0 += pj * (float)vrow[lane];
      acc1 += pj * (float)vrow[lane + 32];
    }
  }
  const float inv = 1.0f / l_run;
  bf16* orow = o + tq * Dm + h * 64;
  orow[lane]      = (bf16)(acc0 * inv);
  orow[lane + 32] = (bf16)(acc1 * inv);
}

// ---------------- MoE gate: softmax, top-2 combine, probs/argmax for LB loss ----------------
__global__ __launch_bounds__(256) void gate_kernel(
    const bf16* __restrict__ n, const float* __restrict__ gw, const float* __restrict__ gb,
    float* __restrict__ comb, float* __restrict__ probs_out, int* __restrict__ amax_out,
    int D, int E)
{
  const int lane = threadIdx.x & 31, wave = threadIdx.x >> 5;
  const int t = blockIdx.x * 8 + wave;
  float a[8] = {0,0,0,0,0,0,0,0};
  const bf16* nr = n + (size_t)t * D;
  for (int d = lane; d < D; d += 32) {
    const float nv = (float)nr[d];
    const float* g = gw + (size_t)d * E;
    #pragma unroll
    for (int e = 0; e < 8; ++e) a[e] += nv * g[e];
  }
  #pragma unroll
  for (int e = 0; e < 8; ++e)
    for (int off = 16; off > 0; off >>= 1) a[e] += __shfl_xor(a[e], off, 32);
  if (lane == 0) {
    float mx = -__builtin_inff();
    #pragma unroll
    for (int e = 0; e < 8; ++e) { a[e] += gb[e]; mx = fmaxf(mx, a[e]); }
    float sum = 0.f;
    #pragma unroll
    for (int e = 0; e < 8; ++e) { a[e] = __expf(a[e] - mx); sum += a[e]; }
    const float inv = 1.0f / sum;
    int i1 = 0; float v1 = -1.f;
    #pragma unroll
    for (int e = 0; e < 8; ++e) {
      a[e] *= inv;
      probs_out[(size_t)t * 8 + e] = a[e];
      if (a[e] > v1) { v1 = a[e]; i1 = e; }
    }
    amax_out[t] = i1;
    int i2 = 0; float v2 = -1.f;
    #pragma unroll
    for (int e = 0; e < 8; ++e) if (e != i1 && a[e] > v2) { v2 = a[e]; i2 = e; }
    const float z = 1.0f / (v1 + v2);
    #pragma unroll
    for (int e = 0; e < 8; ++e) comb[(size_t)t * 8 + e] = 0.f;
    comb[(size_t)t * 8 + i1] = v1 * z;
    comb[(size_t)t * 8 + i2] = v2 * z;
  }
}

// ---------------- deterministic load-balance loss reduction ----------------
__global__ __launch_bounds__(256) void lb_kernel(
    const float* __restrict__ probs, const int* __restrict__ amax,
    float* __restrict__ out_scalar, int T, int L, int E)
{
  __shared__ float red[256];
  const int tid = threadIdx.x;
  float total = 0.f;
  for (int l = 0; l < L; ++l)
    for (int e = 0; e < E; ++e) {
      float fp = 0.f, pp = 0.f;
      for (int t = tid; t < T; t += 256) {
        pp += probs[((size_t)l * T + t) * E + e];
        fp += (amax[l * T + t] == e) ? 1.f : 0.f;
      }
      red[tid] = pp; __syncthreads();
      for (int s = 128; s > 0; s >>= 1) { if (tid < s) red[tid] += red[tid + s]; __syncthreads(); }
      const float P = red[0] / T; __syncthreads();
      red[tid] = fp; __syncthreads();
      for (int s = 128; s > 0; s >>= 1) { if (tid < s) red[tid] += red[tid + s]; __syncthreads(); }
      const float F = red[0] / T; __syncthreads();
      total += 0.01f * 8.0f * F * P;
    }
  if (tid == 0) *out_scalar = total;
}

// ---------------- weight conversion: fp32 -> bf16 ----------------
__global__ void convert_copy(const float* __restrict__ in, bf16* __restrict__ out, size_t ne) {
  const size_t i = (size_t)blockIdx.x * 256 + threadIdx.x;
  if (i < ne) out[i] = (bf16)in[i];
}

// Tiled transpose-convert: in K x N fp32 -> out N x K bf16; both sides coalesced.
__global__ __launch_bounds__(256) void convert_transpose(
    const float* __restrict__ in, bf16* __restrict__ out, int K, int N)
{
  __shared__ float tile[32][33];
  const int bx = blockIdx.x * 32;                // n base
  const int by = blockIdx.y * 32;                // k base
  const int tx = threadIdx.x & 31, ty = threadIdx.x >> 5;
  #pragma unroll
  for (int i = 0; i < 32; i += 8)
    tile[ty + i][tx] = in[(size_t)(by + ty + i) * N + bx + tx];
  __syncthreads();
  #pragma unroll
  for (int i = 0; i < 32; i += 8)
    out[(size_t)(bx + ty + i) * K + by + tx] = (bf16)tile[tx][ty + i];
}

// ---------------- host orchestration ----------------
extern "C" void kernel_launch(void* const* d_in, const int* in_sizes, int n_in,
                              void* d_out, int out_size, void* d_ws, size_t ws_size,
                              hipStream_t stream) {
  (void)in_sizes; (void)n_in; (void)out_size; (void)ws_size;
  const int Bn = 2, S = 1024, T = 2048, D = 1024, H = 16, F = 4096, E = 8, L = 2, V = 32000;
  const int D3 = 3 * D;

  const int*   ids  = (const int*)d_in[0];
  const float* tok  = (const float*)d_in[1];
  const float* pos  = (const float*)d_in[2];
  const float* ln1w = (const float*)d_in[3];
  const float* ln1b = (const float*)d_in[4];
  const float* qkvw = (const float*)d_in[5];
  const float* qkvb = (const float*)d_in[6];
  const float* aow  = (const float*)d_in[7];
  const float* aob  = (const float*)d_in[8];
  const float* ln2w = (const float*)d_in[9];
  const float* ln2b = (const float*)d_in[10];
  const float* gw   = (const float*)d_in[11];
  const float* gb   = (const float*)d_in[12];
  const float* w1   = (const float*)d_in[13];
  const float* b1   = (const float*)d_in[14];
  const float* w2   = (const float*)d_in[15];
  const float* b2   = (const float*)d_in[16];
  const float* flnw = (const float*)d_in[17];
  const float* flnb = (const float*)d_in[18];
  float* out = (float*)d_out;

  char* base = (char*)d_ws; size_t off = 0;
  auto grab = [&](size_t bytes) -> char* {
    char* p = base + off; off += (bytes + 255) & ~(size_t)255; return p;
  };
  float* x     = (float*)grab((size_t)T * D * 4);
  bf16*  n     = (bf16*) grab((size_t)T * D * 2);
  bf16*  qkv   = (bf16*) grab((size_t)T * D3 * 2);
  bf16*  o     = (bf16*) grab((size_t)T * D * 2);
  bf16*  hbuf  = (bf16*) grab((size_t)T * F * 2);
  float* comb  = (float*)grab((size_t)T * E * 4);
  float* probs = (float*)grab((size_t)L * T * E * 4);
  int*   amax  = (int*)  grab((size_t)L * T * 4);
  bf16*  qkvwt = (bf16*) grab((size_t)L * D3 * D * 2);
  bf16*  aowt  = (bf16*) grab((size_t)L * D * D * 2);
  bf16*  w1t   = (bf16*) grab((size_t)F * D * 2);
  bf16*  w2t   = (bf16*) grab((size_t)D * F * 2);
  bf16*  embt  = (bf16*) grab((size_t)V * D * 2);

  // ---- once-per-call weight conversions ----
  {
    const size_t ne = (size_t)V * D;
    convert_copy<<<dim3((unsigned)((ne + 255) / 256)), 256, 0, stream>>>(tok, embt, ne);
  }
  for (int l = 0; l < L; ++l) {
    convert_transpose<<<dim3(D3 / 32, D / 32), 256, 0, stream>>>(
        qkvw + (size_t)l * D * D3, qkvwt + (size_t)l * D3 * D, D, D3);
    convert_transpose<<<dim3(D / 32, D / 32), 256, 0, stream>>>(
        aow + (size_t)l * D * D, aowt + (size_t)l * D * D, D, D);
  }

  embed_kernel<<<dim3(T), 256, 0, stream>>>(ids, tok, pos, x, S, D);

  for (int l = 0; l < L; ++l) {
    // attention block
    layernorm_kernel<<<T, 256, 0, stream>>>(x, ln1w + l * D, ln1b + l * D, n, D);
    gemm_bf16<true, false, true, false, false><<<dim3(D3 / 128, T / 128), 128, 0, stream>>>(
        n, qkvwt + (size_t)l * D3 * D, qkvb + l * D3, nullptr, 0, nullptr, qkv, T, D3, D);
    attn_kernel<<<dim3(Bn * H, S / 8), 256, 0, stream>>>(qkv, o, S, D, H);
    gemm_bf16<true, false, false, true, false><<<dim3(D / 128, T / 128), 128, 0, stream>>>(
        o, aowt + (size_t)l * D * D, aob + l * D, nullptr, 0, x, nullptr, T, D, D);

    // MoE block (dense all-expert compute, combine-weighted accumulate)
    layernorm_kernel<<<T, 256, 0, stream>>>(x, ln2w + l * D, ln2b + l * D, n, D);
    gate_kernel<<<T / 8, 256, 0, stream>>>(
        n, gw + (size_t)l * D * E, gb + l * E, comb,
        probs + (size_t)l * T * E, amax + (size_t)l * T, D, E);
    for (int e = 0; e < E; ++e) {
      convert_transpose<<<dim3(F / 32, D / 32), 256, 0, stream>>>(
          w1 + ((size_t)(l * E + e)) * D * F, w1t, D, F);
      gemm_bf16<true, true, true, false, false><<<dim3(F / 128, T / 128), 128, 0, stream>>>(
          n, w1t, b1 + (size_t)(l * E + e) * F, nullptr, 0, nullptr, hbuf, T, F, D);
      convert_transpose<<<dim3(D / 32, F / 32), 256, 0, stream>>>(
          w2 + ((size_t)(l * E + e)) * F * D, w2t, F, D);
      gemm_bf16<true, false, false, true, true><<<dim3(D / 128, T / 128), 128, 0, stream>>>(
          hbuf, w2t, b2 + (size_t)(l * E + e) * D, comb + e, E, x, nullptr, T, D, F);
    }
  }

  // final LN + tied lm_head
  layernorm_kernel<<<T, 256, 0, stream>>>(x, flnw, flnb, n, D);
  gemm_bf16<false, false, false, false, false><<<dim3(V / 128, T / 128), 128, 0, stream>>>(
      n, embt, nullptr, nullptr, 0, out, nullptr, T, V, D);

  // load-balance loss scalar (deterministic tree reduction)
  lb_kernel<<<1, 256, 0, stream>>>(probs, amax, out + (size_t)T * V, T, L, E);
}
